// LSTM2_70540542870004
// MI455X (gfx1250) — matrix operand order repaired and checked by reading, not compiled, not run on hardware
//
#include <hip/hip_runtime.h>

#define B_SZ  4096
#define T_SZ  128
#define D_SZ  64
#define H_SZ  100
#define G4    400
#define GPAD  416     // padded gate rows (covers 100*3 + 16*6 + 15 = 411)
#define BM    32      // batch rows per workgroup
#define NWAVE 14      // (mt in 0..1) x (j0 in 0..6)
#define NTHR  (NWAVE * 32)

typedef __attribute__((ext_vector_type(16))) __bf16 v16bf;
typedef __attribute__((ext_vector_type(8)))  float  v8f;

union FragAB { v16bf v; uint4 q[2]; };

__device__ __forceinline__ unsigned short f2bf(float x) {
    unsigned int u = __float_as_uint(x);
    u += 0x7FFFu + ((u >> 16) & 1u);   // round-to-nearest-even
    return (unsigned short)(u >> 16);
}
__device__ __forceinline__ float bf2f(unsigned short b) {
    return __uint_as_float((unsigned int)b << 16);
}
__device__ __forceinline__ float sigm_(float x) { return 1.0f / (1.0f + __expf(-x)); }
__device__ __forceinline__ float tanh_(float x) { return 1.0f - 2.0f / (__expf(2.0f * x) + 1.0f); }

// Fused LSTM layer: one workgroup owns BM batch rows across all T steps.
// gates[BM,400] = [x_t | h | 1 | 1] @ [w_ih | w_hh | b_hi | b_lo]^T,
// K padded to KPAD. Bias rides in two constant-1.0 K-columns (hi/lo split ->
// ~fp32-accurate bias, zero extra cost). First K-chunk uses the WMMA inline-0
// C operand, so no accumulator-init movs per step.
// Wave (mt,j0) computes the i/f/g/o 16x16 tiles for batch band mt and hidden
// cols [16*j0, 16*j0+16); cell state c lives in VGPRs for the whole sequence.
// LDS rows use stride KPAD+8 (bf16) => conflict-free ds_load_b128 fragments.
template <int DIN, int KPAD, bool IN_F32, bool SEQ_OUT>
__launch_bounds__(NTHR, 1)
__global__ void lstm_layer_kernel(const float* __restrict__ x_f32,
                                  const unsigned short* __restrict__ x_bf,
                                  const float* __restrict__ w_ih,   // [400, DIN]
                                  const float* __restrict__ w_hh,   // [400, 100]
                                  const float* __restrict__ b_ih,   // [400]
                                  const float* __restrict__ b_hh,   // [400]
                                  unsigned short* __restrict__ seq_out, // bf16 [B,T,100]
                                  float* __restrict__ last_out)         // f32  [B,100]
{
    constexpr int STR = KPAD + 8;               // bf16 elements per LDS row
    constexpr int NX  = (BM * DIN + NTHR - 1) / NTHR;  // x elems per thread
    constexpr int CB0 = DIN + H_SZ;             // bias-hi column (was zero pad)
    constexpr int CB1 = DIN + H_SZ + 1;         // bias-lo column

    extern __shared__ char smem[];
    unsigned short* W_lds = (unsigned short*)smem;              // [GPAD][STR]
    unsigned short* A_lds = W_lds + GPAD * STR;                 // [BM][STR]

    const int tid  = threadIdx.x;
    const int lane = tid & 31;
    const int wave = tid >> 5;
    const int mt   = wave / 7;       // batch band (0..1)
    const int j0   = wave % 7;       // hidden-col tile (0..6)
    const int half = lane >> 4;
    const int ln   = lane & 15;
    const int b0   = blockIdx.x * BM;

    // ---- one-time: weights (concat + bias cols + zero pad) -> LDS bf16 ----
    for (int idx = tid; idx < GPAD * STR; idx += NTHR) {
        int row = idx / STR, k = idx - row * STR;
        unsigned short v = 0;
        if (row < G4) {
            if (k < DIN) {
                v = f2bf(w_ih[row * DIN + k]);
            } else if (k < DIN + H_SZ) {
                v = f2bf(w_hh[row * H_SZ + (k - DIN)]);
            } else if (k == CB0 || k == CB1) {
                float bfull = b_ih[row] + b_hh[row];
                unsigned short hi = f2bf(bfull);
                v = (k == CB0) ? hi : f2bf(bfull - bf2f(hi));
            }
        }
        W_lds[idx] = v;
    }
    // A staging: h0 = 0, pad = 0, constant 1.0 in the two bias columns
    for (int idx = tid; idx < BM * STR; idx += NTHR) {
        int k = idx % STR;
        A_lds[idx] = (k == CB0 || k == CB1) ? (unsigned short)0x3F80 : (unsigned short)0;
    }

    const uint4* A4 = reinterpret_cast<const uint4*>(A_lds);
    const uint4* W4 = reinterpret_cast<const uint4*>(W_lds);

    // ---- prefetch x_0 into registers (double-buffered staging) ----
    unsigned short xreg[NX];
    #pragma unroll
    for (int i = 0; i < NX; ++i) {
        int idx = tid + i * NTHR;
        if (idx < BM * DIN) {
            int m = idx / DIN, k = idx - m * DIN;
            size_t gix = ((size_t)(b0 + m) * T_SZ + 0) * DIN + k;
            xreg[i] = IN_F32 ? f2bf(x_f32[gix]) : x_bf[gix];
        }
    }
    __syncthreads();   // weights/A_lds init visible

    int colw_dw[4];
    #pragma unroll
    for (int g = 0; g < 4; ++g)
        colw_dw[g] = (H_SZ * g + 16 * j0 + ln) * (STR / 2);
    const int a_row_dw = (16 * mt + ln) * (STR / 2);

    v8f c_acc;
    #pragma unroll
    for (int i = 0; i < 8; ++i) c_acc[i] = 0.0f;

    for (int t = 0; t < T_SZ; ++t) {
        // ---- commit staged x_t regs into A_lds cols [0, DIN) ----
        #pragma unroll
        for (int i = 0; i < NX; ++i) {
            int idx = tid + i * NTHR;
            if (idx < BM * DIN) {
                int m = idx / DIN, k = idx - m * DIN;
                A_lds[m * STR + k] = xreg[i];
            }
        }
        __syncthreads();   // x/h writes visible before GEMM reads

        // ---- prefetch x_{t+1}: latency overlaps the WMMA K-loop ----
        if (t + 1 < T_SZ) {
            #pragma unroll
            for (int i = 0; i < NX; ++i) {
                int idx = tid + i * NTHR;
                if (idx < BM * DIN) {
                    int m = idx / DIN, k = idx - m * DIN;
                    size_t gix = ((size_t)(b0 + m) * T_SZ + (t + 1)) * DIN + k;
                    xreg[i] = IN_F32 ? f2bf(x_f32[gix]) : x_bf[gix];
                }
            }
        }

        // ---- concat-K GEMM: 4 gate tiles, K = KPAD in chunks of 32 ----
        v8f accs[4];
        {   // kc = 0: C is inline zero -> no accumulator-init movs
            FragAB fa;
            int ba = a_row_dw + 4 * half;
            fa.q[0] = A4[ba >> 2];
            fa.q[1] = A4[(ba >> 2) + 2];
            const v8f zc = {};
            #pragma unroll
            for (int g = 0; g < 4; ++g) {
                FragAB fb;
                int bw = colw_dw[g] + 4 * half;
                fb.q[0] = W4[bw >> 2];
                fb.q[1] = W4[(bw >> 2) + 2];
                accs[g] = __builtin_amdgcn_wmma_f32_16x16x32_bf16(
                    false, fa.v, false, fb.v, (short)0, zc, false, false);
            }
        }
        #pragma unroll
        for (int kc = 1; kc < KPAD / 32; ++kc) {
            FragAB fa;
            int ba = a_row_dw + 16 * kc + 4 * half;        // A frag (16x32 bf16)
            fa.q[0] = A4[ba >> 2];
            fa.q[1] = A4[(ba >> 2) + 2];
            #pragma unroll
            for (int g = 0; g < 4; ++g) {
                FragAB fb;
                int bw = colw_dw[g] + 16 * kc + 4 * half;  // B frag (32x16 bf16)
                fb.q[0] = W4[bw >> 2];
                fb.q[1] = W4[(bw >> 2) + 2];
                accs[g] = __builtin_amdgcn_wmma_f32_16x16x32_bf16(
                    false, fa.v, false, fb.v, (short)0, accs[g], false, false);
            }
        }
        __syncthreads();   // all A_lds reads done before h writeback

        // ---- gate nonlinearities + cell update (registers only) ----
        const int  col   = 16 * j0 + ln;
        const bool colok = col < H_SZ;
        #pragma unroll
        for (int v = 0; v < 8; ++v) {
            float ii = sigm_(accs[0][v]);
            float ff = sigm_(accs[1][v]);
            float gg = tanh_(accs[2][v]);
            float oo = sigm_(accs[3][v]);
            float cn = ff * c_acc[v] + ii * gg;
            c_acc[v] = cn;
            float hv = oo * tanh_(cn);
            int m = 16 * mt + v + 8 * half;   // C/D layout: m = vgpr + 8*half
            if (colok) {
                A_lds[m * STR + DIN + col] = f2bf(hv);     // recurrent feedback (raw h)
                float hr = fmaxf(hv, 0.0f);                // relu only on layer output
                if (SEQ_OUT) {
                    seq_out[((size_t)(b0 + m) * T_SZ + t) * H_SZ + col] = f2bf(hr);
                } else if (t == T_SZ - 1) {
                    last_out[(b0 + m) * H_SZ + col] = hr;
                }
            }
        }
        // x writes (cols < DIN) and h writes (cols in [DIN, DIN+100)) are
        // disjoint from each other and from the constant bias columns; the
        // barrier after x-commit fences all writes against the next GEMM.
    }
}

// logits = h_last @ w_fc^T + b_fc ; softmax over 8 classes. Tiny: 1 row/thread.
__global__ void fc_softmax_kernel(const float* __restrict__ h,     // [B,100]
                                  const float* __restrict__ w_fc,  // [8,100]
                                  const float* __restrict__ b_fc,  // [8]
                                  float* __restrict__ out)         // [B,8]
{
    __shared__ float wf[8 * H_SZ];
    __shared__ float bf[8];
    for (int i = threadIdx.x; i < 8 * H_SZ; i += blockDim.x) wf[i] = w_fc[i];
    if (threadIdx.x < 8) bf[threadIdx.x] = b_fc[threadIdx.x];
    __syncthreads();

    int b = blockIdx.x * blockDim.x + threadIdx.x;
    if (b >= B_SZ) return;

    float acc[8];
    #pragma unroll
    for (int c = 0; c < 8; ++c) acc[c] = bf[c];
    for (int k = 0; k < H_SZ; ++k) {
        float hv = h[b * H_SZ + k];
        #pragma unroll
        for (int c = 0; c < 8; ++c) acc[c] += hv * wf[c * H_SZ + k];
    }
    float mx = acc[0];
    #pragma unroll
    for (int c = 1; c < 8; ++c) mx = fmaxf(mx, acc[c]);
    float s = 0.0f;
    #pragma unroll
    for (int c = 0; c < 8; ++c) { acc[c] = __expf(acc[c] - mx); s += acc[c]; }
    float inv = 1.0f / s;
    #pragma unroll
    for (int c = 0; c < 8; ++c) out[b * 8 + c] = acc[c] * inv;
}

extern "C" void kernel_launch(void* const* d_in, const int* in_sizes, int n_in,
                              void* d_out, int out_size, void* d_ws, size_t ws_size,
                              hipStream_t stream)
{
    const float* x     = (const float*)d_in[0];
    const float* w_ih1 = (const float*)d_in[1];
    const float* w_hh1 = (const float*)d_in[2];
    const float* b_ih1 = (const float*)d_in[3];
    const float* b_hh1 = (const float*)d_in[4];
    const float* w_ih2 = (const float*)d_in[5];
    const float* w_hh2 = (const float*)d_in[6];
    const float* b_ih2 = (const float*)d_in[7];
    const float* b_hh2 = (const float*)d_in[8];
    const float* w_fc  = (const float*)d_in[9];
    const float* b_fc  = (const float*)d_in[10];

    // workspace: h1 as bf16 [B,T,100] (~105 MB, fits in 192 MB L2),
    // then last-step h2 as f32 [B,100]
    unsigned short* h1 = (unsigned short*)d_ws;
    size_t h1_bytes = (size_t)B_SZ * T_SZ * H_SZ * sizeof(unsigned short);
    float* h2_last = (float*)((char*)d_ws + h1_bytes);

    constexpr int KPAD1 = 192;  // 64 + 100 (+2 bias cols) -> 192
    constexpr int KPAD2 = 224;  // 100 + 100 (+2 bias cols) -> 224
    size_t sm1 = (size_t)(GPAD + BM) * (KPAD1 + 8) * 2;  // ~179 KB
    size_t sm2 = (size_t)(GPAD + BM) * (KPAD2 + 8) * 2;  // ~208 KB (<320KB/WGP)

    dim3 grid(B_SZ / BM);  // 128 workgroups, 14 waves each

    lstm_layer_kernel<D_SZ, KPAD1, true, true>
        <<<grid, NTHR, sm1, stream>>>(x, nullptr, w_ih1, w_hh1, b_ih1, b_hh1,
                                      h1, nullptr);

    lstm_layer_kernel<H_SZ, KPAD2, false, false>
        <<<grid, NTHR, sm2, stream>>>(nullptr, h1, w_ih2, w_hh2, b_ih2, b_hh2,
                                      nullptr, h2_last);

    fc_softmax_kernel<<<(B_SZ + 255) / 256, 256, 0, stream>>>(h2_last, w_fc, b_fc,
                                                              (float*)d_out);
}
